// AxialSelfAttention2d_11931419148695
// MI455X (gfx1250) — matrix operand order repaired and checked by reading, not compiled, not run on hardware
//
#include <hip/hip_runtime.h>

typedef __bf16 bf16;
typedef __bf16 v4bf  __attribute__((ext_vector_type(4)));
typedef __bf16 v8bf  __attribute__((ext_vector_type(8)));
typedef __bf16 v16bf __attribute__((ext_vector_type(16)));
typedef float  v8f   __attribute__((ext_vector_type(8)));

#define DEV __device__ __forceinline__

// ---------------------------------------------------------------------------
// WMMA fragment helpers (gfx1250 wave32 layouts, CDNA5 ISA §7.12.2)
//
// A (16x32 bf16, MxK): lane holds row M = lane&15.
//   elements 0..7  -> K = 8*(lane>=16) + i        (contiguous 8, 16B)
//   elements 8..15 -> K = 16 + 8*(lane>=16) + i-8 (contiguous 8, 16B)
// B (32x16 bf16, KxN): lane holds col N = lane&15.
//   elements 0..7  -> K = 8*(lane>=16) + i; 8..15 -> K = 16 + 8*(lane>=16) + i-8
// C/D (16x16 f32): VGPR r at lane l holds M = r + 8*(l>>4), N = l&15.
// ---------------------------------------------------------------------------

DEV v16bf concat8(v8bf lo, v8bf hi) {
  return __builtin_shufflevector(lo, hi, 0,1,2,3,4,5,6,7,8,9,10,11,12,13,14,15);
}

DEV v16bf load_a_frag(const bf16* base, int row_stride, int lane) {
  const bf16* p = base + (lane & 15) * row_stride + ((lane >> 4) << 3);
  v8bf lo = *(const v8bf*)p;
  v8bf hi = *(const v8bf*)(p + 16);
  return concat8(lo, hi);
}

DEV v16bf load_b_frag(const bf16* base, int row_stride, int lane) {
  const bf16* p = base + (lane & 15) * row_stride + ((lane >> 4) << 3);
  v8bf lo = *(const v8bf*)p;
  v8bf hi = *(const v8bf*)(p + 16);
  return concat8(lo, hi);
}

DEV v8f wmma_bf16(v16bf a, v16bf b, v8f c) {
  return __builtin_amdgcn_wmma_f32_16x16x32_bf16(false, a, false, b, (short)0, c,
                                                 false, false);
}

// Async memory->LDS dword copy (CDNA5 ISA, ASYNCcnt).
// GV addressing: VDST vgpr = LDS byte offset, VADDR = 64-bit global address.
// LDS offset = low 32 bits of the generic shared pointer (ISA §10.2: flat
// shared addresses carry the wave-relative LDS offset in addr[31:0]).
DEV void async_copy_b32(unsigned lds_off, const bf16* gptr) {
  asm volatile("global_load_async_to_lds_b32 %0, %1, off"
               :
               : "v"(lds_off), "v"((unsigned long long)(uintptr_t)gptr)
               : "memory");
}
DEV void wait_async0() { asm volatile("s_wait_asynccnt 0" ::: "memory"); }

// reductions across the 16-lane half (xor masks 1..8 stay within the half)
DEV float half16_max(float v) {
  v = fmaxf(v, __shfl_xor(v, 1));
  v = fmaxf(v, __shfl_xor(v, 2));
  v = fmaxf(v, __shfl_xor(v, 4));
  v = fmaxf(v, __shfl_xor(v, 8));
  return v;
}
DEV float half16_sum(float v) {
  v += __shfl_xor(v, 1); v += __shfl_xor(v, 2);
  v += __shfl_xor(v, 4); v += __shfl_xor(v, 8);
  return v;
}

// ---------------------------------------------------------------------------
// f32 -> bf16 pack (vectorized x4)
// ---------------------------------------------------------------------------
__global__ __launch_bounds__(256)
void cvt_f32_bf16_k(const float* __restrict__ in, bf16* __restrict__ out, int n4) {
  int i = blockIdx.x * blockDim.x + threadIdx.x;
  if (i < n4) {
    float4 f = ((const float4*)in)[i];
    v4bf o;
    o.x = (bf16)f.x; o.y = (bf16)f.y; o.z = (bf16)f.z; o.w = (bf16)f.w;
    *(v4bf*)(out + (size_t)i * 4) = o;
  }
}

// ---------------------------------------------------------------------------
// QKV projection: out[M,1536] = A[M,512] @ W[1536,512]^T + bias  (bf16 in/out,
// f32 accumulate). Block 256 thr = 8 waves; wave -> 32x64 tile; grid (24,128).
// A and W both live in L2 (33.5 MB), so direct-fragment loads are fed from L2.
// ---------------------------------------------------------------------------
__global__ __launch_bounds__(256)
void gemm_qkv_k(const bf16* __restrict__ A, const bf16* __restrict__ W,
                const float* __restrict__ bias, bf16* __restrict__ out) {
  const int lane  = threadIdx.x & 31;
  const int wave  = threadIdx.x >> 5;
  const int nbase = blockIdx.x * 64;
  const int mbase = blockIdx.y * 256 + wave * 32;

  v8f acc[2][4];
  for (int mt = 0; mt < 2; ++mt)
    for (int nt = 0; nt < 4; ++nt) acc[mt][nt] = (v8f)0.0f;

  for (int kc = 0; kc < 512; kc += 32) {
    v16bf a0 = load_a_frag(A + (size_t)mbase * 512 + kc, 512, lane);
    v16bf a1 = load_a_frag(A + (size_t)(mbase + 16) * 512 + kc, 512, lane);
    for (int nt = 0; nt < 4; ++nt) {
      v16bf b = load_b_frag(W + (size_t)(nbase + nt * 16) * 512 + kc, 512, lane);
      acc[0][nt] = wmma_bf16(a0, b, acc[0][nt]);
      acc[1][nt] = wmma_bf16(a1, b, acc[1][nt]);
    }
  }

  const int mloc = (lane >> 4) << 3, nloc = lane & 15;
  for (int nt = 0; nt < 4; ++nt) {
    float bv = bias[nbase + nt * 16 + nloc];
    for (int mt = 0; mt < 2; ++mt)
      for (int r = 0; r < 8; ++r) {
        float v = acc[mt][nt][r] + bv;
        out[(size_t)(mbase + mt * 16 + mloc + r) * 1536 + nbase + nt * 16 + nloc] =
            (bf16)v;
      }
  }
}

// ---------------------------------------------------------------------------
// Fused attention head: one workgroup per (batch-slice, head).
// K/V head-tiles staged memory->LDS with global_load_async_to_lds_b32
// (ASYNCcnt path, no VGPR round-trip); V then transposed LDS->LDS so P@V is
// NT-shaped for WMMA. Flash-style online softmax over 32-key chunks; P is
// re-fragmented (C-layout -> A-layout) through wave-local LDS that overlays
// the V staging scratch (disjoint lifetimes, separated by a barrier).
//  ROW:  SEQ=256, 8 waves x 32 query rows   (grid = B*S*H = 1024, 96KB LDS)
//  COL:  SEQ=64,  4 waves x 16 query rows   (grid = B*L*H = 4096, 24KB LDS)
// ---------------------------------------------------------------------------
template <int SEQ, int NWAVES, int MT, bool COL>
__global__ __launch_bounds__(NWAVES * 32)
void attn_k(const bf16* __restrict__ qkv, float* __restrict__ attn) {
  const int lane = threadIdx.x & 31;
  const int wave = threadIdx.x >> 5;

  int h, base, stride;
  if (!COL) {                       // (b,s) slice, attend over L
    int bi = blockIdx.x; h = bi & 7; int bs = bi >> 3;
    base = bs * 256; stride = 1;
  } else {                          // (b,l) slice, attend over S
    int bi = blockIdx.x; h = bi & 7; int t = bi >> 3;
    int l = t & 255; int b = t >> 8;
    base = b * (64 * 256) + l; stride = 256;
  }

  constexpr int PSZ = NWAVES * MT * 16 * 32;
  constexpr int SCR = (SEQ * 64 > PSZ) ? SEQ * 64 : PSZ;
  __shared__ __align__(16) bf16 Kbuf[SEQ * 64];   // [key j][c] (N-major)
  __shared__ __align__(16) bf16 Vt[64 * SEQ];     // [c][key j] (N-major)
  __shared__ __align__(16) bf16 Scratch[SCR];     // V staging, then P buffers

  // --- async stage K and V tiles (row-major [j][c]), 1 dword = 2 bf16/lane ---
  for (int idx = threadIdx.x; idx < SEQ * 32; idx += NWAVES * 32) {
    int j = idx >> 5, cp = (idx & 31) * 2;
    const bf16* qr = qkv + (size_t)(base + j * stride) * 1536 + h * 64 + cp;
    async_copy_b32((unsigned)(uintptr_t)&Kbuf[j * 64 + cp],    qr + 512);
    async_copy_b32((unsigned)(uintptr_t)&Scratch[j * 64 + cp], qr + 1024);
  }
  wait_async0();
  __syncthreads();
  // --- transpose V: Scratch[j][c] -> Vt[c][j] ---
  for (int idx = threadIdx.x; idx < SEQ * 64; idx += NWAVES * 32) {
    int j = idx >> 6, c = idx & 63;
    Vt[c * SEQ + j] = Scratch[j * 64 + c];
  }
  __syncthreads();

  const int m0 = wave * (MT * 16);
  v16bf qf[MT][2];
  for (int mt = 0; mt < MT; ++mt)
    for (int kc = 0; kc < 2; ++kc)
      qf[mt][kc] = load_a_frag(
          qkv + (size_t)(base + (m0 + mt * 16) * stride) * 1536 + h * 64 + kc * 32,
          stride * 1536, lane);

  v8f O[MT][4];
  float mrow[MT][8], lrow[MT][8];
  for (int mt = 0; mt < MT; ++mt) {
    for (int vc = 0; vc < 4; ++vc) O[mt][vc] = (v8f)0.0f;
    for (int r = 0; r < 8; ++r) { mrow[mt][r] = -1e30f; lrow[mt][r] = 0.0f; }
  }
  const int mloc = (lane >> 4) << 3, nloc = lane & 15;
  bf16* pwave = &Scratch[wave * (MT * 16 * 32)];

  for (int k0 = 0; k0 < SEQ; k0 += 32) {
    for (int mt = 0; mt < MT; ++mt) {
      v8f s0 = (v8f)0.0f, s1 = (v8f)0.0f;   // scores vs keys k0..k0+15, +16..31
      for (int kc = 0; kc < 2; ++kc) {
        v16bf bk0 = load_b_frag(&Kbuf[(k0 +  0) * 64 + kc * 32], 64, lane);
        v16bf bk1 = load_b_frag(&Kbuf[(k0 + 16) * 64 + kc * 32], 64, lane);
        s0 = wmma_bf16(qf[mt][kc], bk0, s0);
        s1 = wmma_bf16(qf[mt][kc], bk1, s1);
      }
      float scale[8];
      v8f p0, p1;
      for (int r = 0; r < 8; ++r) {
        float tm = half16_max(fmaxf(s0[r], s1[r]));
        float mn = fmaxf(mrow[mt][r], tm);
        scale[r] = __expf(mrow[mt][r] - mn);
        mrow[mt][r] = mn;
        float e0 = __expf(s0[r] - mn);
        float e1 = __expf(s1[r] - mn);
        p0[r] = e0; p1[r] = e1;
        lrow[mt][r] = lrow[mt][r] * scale[r] + half16_sum(e0 + e1);
      }
      for (int vc = 0; vc < 4; ++vc)
        for (int r = 0; r < 8; ++r) O[mt][vc][r] *= scale[r];
      // C-layout -> row-major [16][32] bf16 in wave-local LDS
      bf16* pb = pwave + mt * 16 * 32;
      for (int r = 0; r < 8; ++r) {
        pb[(mloc + r) * 32 + nloc]      = (bf16)p0[r];
        pb[(mloc + r) * 32 + 16 + nloc] = (bf16)p1[r];
      }
    }
    asm volatile("s_wait_dscnt 0" ::: "memory");   // wave-local LDS RAW
    for (int mt = 0; mt < MT; ++mt) {
      v16bf pa = load_a_frag(pwave + mt * 16 * 32, 32, lane);
      for (int vc = 0; vc < 4; ++vc) {
        v16bf vb = load_b_frag(&Vt[(vc * 16) * SEQ + k0], SEQ, lane);
        O[mt][vc] = wmma_bf16(pa, vb, O[mt][vc]);
      }
    }
  }

  for (int mt = 0; mt < MT; ++mt) {
    float inv[8];
    for (int r = 0; r < 8; ++r) inv[r] = 1.0f / lrow[mt][r];
    for (int vc = 0; vc < 4; ++vc)
      for (int r = 0; r < 8; ++r) {
        int q = m0 + mt * 16 + mloc + r;
        size_t n = (size_t)(base + q * stride);
        attn[n * 512 + h * 64 + vc * 16 + nloc] = O[mt][vc][r] * inv[r];
      }
  }
}

// ---------------------------------------------------------------------------
// Residual + LayerNorm over D=512; one wave per row, 16 f32 per lane.
// Writes f32 result + optional fused bf16 copy (next stage's GEMM operand).
// ---------------------------------------------------------------------------
__global__ __launch_bounds__(256)
void add_ln_k(const float* __restrict__ xin, const float* __restrict__ attn,
              const float* __restrict__ g, const float* __restrict__ beta,
              float* __restrict__ out, bf16* __restrict__ out_bf, int write_bf) {
  int lane = threadIdx.x & 31, wave = threadIdx.x >> 5;
  size_t row = (size_t)blockIdx.x * 8 + wave;
  const float* xr = xin + row * 512;
  const float* ar = attn + row * 512;
  float v[16], s = 0.0f, ss = 0.0f;
  for (int i = 0; i < 16; ++i) {
    int idx = i * 32 + lane;
    float t = xr[idx] + ar[idx];
    v[i] = t; s += t; ss += t * t;
  }
  for (int off = 1; off < 32; off <<= 1) {
    s += __shfl_xor(s, off);
    ss += __shfl_xor(ss, off);
  }
  float mean = s * (1.0f / 512.0f);
  float rstd = rsqrtf(ss * (1.0f / 512.0f) - mean * mean + 1e-5f);
  for (int i = 0; i < 16; ++i) {
    int idx = i * 32 + lane;
    float t = (v[i] - mean) * rstd * g[idx] + beta[idx];
    out[row * 512 + idx] = t;
    if (write_bf) out_bf[row * 512 + idx] = (bf16)t;
  }
}

// ---------------------------------------------------------------------------
// Host side
// ---------------------------------------------------------------------------
extern "C" void kernel_launch(void* const* d_in, const int* in_sizes, int n_in,
                              void* d_out, int out_size, void* d_ws, size_t ws_size,
                              hipStream_t stream) {
  (void)in_sizes; (void)n_in; (void)out_size; (void)ws_size;
  const float* x     = (const float*)d_in[0];
  const float* w_row = (const float*)d_in[1];
  const float* b_row = (const float*)d_in[2];
  const float* w_col = (const float*)d_in[3];
  const float* b_col = (const float*)d_in[4];
  const float* g1    = (const float*)d_in[5];
  const float* be1   = (const float*)d_in[6];
  const float* g2    = (const float*)d_in[7];
  const float* be2   = (const float*)d_in[8];
  float* out = (float*)d_out;

  const size_t NROW = 32768;  // B*S*L
  char* ws = (char*)d_ws;
  bf16*  xb   = (bf16*)ws;                                   // 32  MiB act (bf16)
  bf16*  wb   = (bf16*)(ws + NROW * 512 * 2);                // 1.5 MiB weights
  bf16*  qkv  = (bf16*)(ws + NROW * 512 * 2 + 1536 * 512 * 2);        // 96 MiB
  float* attn = (float*)(ws + NROW * 512 * 2 + 1536 * 512 * 2 +
                         NROW * 1536 * 2);                   // 64 MiB f32

  dim3 b256(256), b128(128);
  const int nx4 = (int)(NROW * 512 / 4);
  const int nw4 = 1536 * 512 / 4;

  // ---- stage 1: row attention over L ----
  cvt_f32_bf16_k<<<(nx4 + 255) / 256, b256, 0, stream>>>(x, xb, nx4);
  cvt_f32_bf16_k<<<(nw4 + 255) / 256, b256, 0, stream>>>(w_row, wb, nw4);
  gemm_qkv_k<<<dim3(24, 128), b256, 0, stream>>>(xb, wb, b_row, qkv);
  attn_k<256, 8, 2, false><<<1024, b256, 0, stream>>>(qkv, attn);
  add_ln_k<<<4096, b256, 0, stream>>>(x, attn, g1, be1, out, xb, 1);

  // ---- stage 2: column attention over S ----
  cvt_f32_bf16_k<<<(nw4 + 255) / 256, b256, 0, stream>>>(w_col, wb, nw4);
  gemm_qkv_k<<<dim3(24, 128), b256, 0, stream>>>(xb, wb, b_col, qkv);
  attn_k<64, 4, 1, true><<<4096, b128, 0, stream>>>(qkv, attn);
  add_ln_k<<<4096, b256, 0, stream>>>(out, attn, g2, be2, out, xb, 0);
}
